// SimpleProjectNetwork_23313082483150
// MI455X (gfx1250) — matrix operand order repaired and checked by reading, not compiled, not run on hardware
//
#include <hip/hip_runtime.h>
#include <math.h>

typedef __attribute__((ext_vector_type(16))) __bf16 v16bf;
typedef __attribute__((ext_vector_type(8)))  float  v8f;

#define CCH 64     // channels
#define K2  128    // concat width
// per-layer prepped B: 16 fragments (kc,t) x 32 lanes x 16 bf16 = 8192 bf16 = 16KB
#define FRAG_ELEMS 512
#define LAYER_FRAG_ELEMS (16 * FRAG_ELEMS)

// ---------------- zero fill (b128 stores) ----------------
__global__ void zero_kernel(float4* __restrict__ p, int n4) {
    int i = blockIdx.x * blockDim.x + threadIdx.x;
    if (i < n4) p[i] = make_float4(0.f, 0.f, 0.f, 0.f);
}

// ---------------- one-time W -> bf16 fragment repack ----------------
// fragment id gw = l*16 + kc*4 + t ; one wave per fragment.
// lane lr<16 holds N = t*16+lr ; element e = B[kc*32 + hi*16 + e][N] = W[l][N][k]
__global__ __launch_bounds__(256) void prep_w_kernel(
        const float* __restrict__ W,      // [L, 64, 128]
        __bf16*      __restrict__ Wfrag,  // [L*16*512]
        int nfrag) {
    int gw   = (blockIdx.x * blockDim.x + threadIdx.x) >> 5;
    int lane = threadIdx.x & 31;
    if (gw >= nfrag) return;
    int l  = gw >> 4;
    int kc = (gw >> 2) & 3;
    int t  = gw & 3;
    int lr = lane & 15;
    int hi = lane >> 4;
    const float* wrow = W + ((size_t)l * CCH + (t * 16 + lr)) * K2 + (kc * 32 + hi * 16);
    __bf16* dst = Wfrag + (size_t)gw * FRAG_ELEMS + lane * 16;
#pragma unroll
    for (int e = 0; e < 16; ++e) dst[e] = (__bf16)wrow[e];
}

// ---------------- edge gather + scatter-add ----------------
// one wave32 per edge; each lane handles 2 channels (float2, 256B coalesced row)
__global__ __launch_bounds__(256) void scatter_kernel(
        const float* __restrict__ h,      // [N, 64]
        const int*   __restrict__ ei,     // [2, E] flat (row0=src, row1=dst)
        const float* __restrict__ ew,     // [E]
        float*       __restrict__ aggr,   // [N, 64]
        int E) {
    int wid  = (blockIdx.x * blockDim.x + threadIdx.x) >> 5;
    int lane = threadIdx.x & 31;
    if (wid >= E) return;
    int   src = ei[wid];
    int   dst = ei[E + wid];
    float w   = ew[wid];
    const float2* hrow = (const float2*)(h + (size_t)src * CCH);
    float2 v = hrow[lane];
    float* arow = aggr + (size_t)dst * CCH + lane * 2;
    // lowers to global_atomic_add_f32 (L2 atomic units, everything L2-resident)
    unsafeAtomicAdd(arow,     v.x * w);
    unsafeAtomicAdd(arow + 1, v.y * w);
}

// fast tanh via hardware transcendentals: exact +/-1 saturation,
// ~1e-7 mid-range error (far below the bf16 GEMM noise floor)
__device__ __forceinline__ float tanh_fast(float x) {
    float e = __expf(2.0f * x);                       // v_exp_f32
    return 1.0f - 2.0f * __builtin_amdgcn_rcpf(e + 1.0f);  // v_rcp_f32
}

// ---------------- fused concat-GEMM + bias + tanh via WMMA ----------------
// out[N,64] = tanh( concat(aggr,h)[N,128] @ W^T + b )
// one wave computes a 16-row x 64-col tile: 4 v8f accumulators,
// K swept in 4 chunks of 32 with v_wmma_f32_16x16x32_bf16.
// B fragments are pre-converted bf16, staged once per block into LDS.
__global__ __launch_bounds__(256) void gemm_tanh_kernel(
        const float*  __restrict__ aggr,    // [N, 64]  (K = 0..63 of concat)
        const float*  __restrict__ hin,     // [N, 64]  (K = 64..127 of concat)
        const __bf16* __restrict__ WfragL,  // [16*512] this layer's fragments
        const float*  __restrict__ bl,      // [64]
        float*        __restrict__ out,     // [N, 64]
        int N) {
    __shared__ __attribute__((aligned(32))) __bf16 sB[LAYER_FRAG_ELEMS]; // 16KB

    // cooperative 16KB copy: 256 threads x 4 x 16B
    {
        const float4* src = (const float4*)WfragL;
        float4*       dst = (float4*)sB;
        int tid = threadIdx.x;
#pragma unroll
        for (int j = 0; j < 4; ++j) dst[tid + j * 256] = src[tid + j * 256];
    }
    __syncthreads();   // before any divergent exit

    const int lane    = threadIdx.x & 31;
    const int wave    = threadIdx.x >> 5;
    const int rowTile = blockIdx.x * 8 + wave;
    const int rowBase = rowTile * 16;
    if (rowBase >= N) return;   // safe: no further barriers

    const int lr = lane & 15;   // lane-in-halfwave
    const int hi = lane >> 4;   // 0 or 1

    // accumulators seeded with bias: c[t] element r -> row (rowBase+hi*8+r), col t*16+lr
    v8f c[4];
#pragma unroll
    for (int t = 0; t < 4; ++t) {
        float bv = bl[t * 16 + lr];
#pragma unroll
        for (int r = 0; r < 8; ++r) c[t][r] = bv;
    }

    // A-fragment row for this lane (clamped so tail tiles keep EXEC all-1s)
    int rowA = rowBase + lr;
    if (rowA >= N) rowA = N - 1;

#pragma unroll
    for (int kc = 0; kc < 4; ++kc) {
        const int k0 = kc * 32;

        // ---- A fragment (contiguous K order chosen by this code, not the ISA table order):
        // lane<16: K = k0+{0..15}; lane>=16: K = k0+{16..31} (matches B fragment order)
        const int bk = k0 + hi * 16;
        const float* arow = (k0 < CCH)
                          ? (aggr + (size_t)rowA * CCH + bk)
                          : (hin  + (size_t)rowA * CCH + (bk - CCH));
        v16bf a;
#pragma unroll
        for (int e = 0; e < 8; ++e) a[e]     = (__bf16)arow[e];
#pragma unroll
        for (int e = 0; e < 8; ++e) a[8 + e] = (__bf16)arow[8 + e];

        // ---- B fragments straight from LDS (already bf16, already laid out)
#pragma unroll
        for (int t = 0; t < 4; ++t) {
            v16bf bf = *(const v16bf*)(sB + ((kc * 4 + t) * FRAG_ELEMS + lane * 16));
            c[t] = __builtin_amdgcn_wmma_f32_16x16x32_bf16(
                       /*neg_a=*/false, a, /*neg_b=*/false, bf,
                       /*c_mod=*/(short)0, c[t],
                       /*reuse_a=*/false, /*reuse_b=*/false);
        }
    }

    // ---- tanh + store (C/D layout: VGPR r -> row r (lanes 0-15) / row 8+r)
#pragma unroll
    for (int t = 0; t < 4; ++t) {
        const int col = t * 16 + lr;
#pragma unroll
        for (int r = 0; r < 8; ++r) {
            int row = rowBase + hi * 8 + r;
            if (row < N) out[(size_t)row * CCH + col] = tanh_fast(c[t][r]);
        }
    }
}

extern "C" void kernel_launch(void* const* d_in, const int* in_sizes, int n_in,
                              void* d_out, int out_size, void* d_ws, size_t ws_size,
                              hipStream_t stream) {
    const float* x  = (const float*)d_in[0];   // [N, 64]
    const int*   ei = (const int*)  d_in[1];   // [2, E]
    const float* ew = (const float*)d_in[2];   // [E]
    const float* W  = (const float*)d_in[3];   // [L, 64, 128]
    const float* b  = (const float*)d_in[4];   // [L, 64]

    const int N = in_sizes[0] / CCH;
    const int E = in_sizes[2];
    const int L = in_sizes[4] / CCH;

    float*  aggr  = (float*)d_ws;
    float*  hA    = aggr + (size_t)N * CCH;
    float*  hB    = hA   + (size_t)N * CCH;
    __bf16* Wfrag = (__bf16*)(hB + (size_t)N * CCH);   // L*16*512 bf16 (128KB)
    float*  buf[2] = { hA, hB };

    const int nAggr      = N * CCH;
    const int zeroBlocks = (nAggr / 4 + 255) / 256;
    const int scatBlocks = (E + 7) / 8;                // 8 waves/block, 1 edge/wave
    const int rowTiles   = (N + 15) / 16;
    const int gemmBlocks = (rowTiles + 7) / 8;         // 8 waves/block, 16 rows/wave
    const int nfrag      = L * 16;

    // one-time W repack to bf16 WMMA-fragment order
    prep_w_kernel<<<(nfrag * 32 + 255) / 256, 256, 0, stream>>>(W, Wfrag, nfrag);

    for (int l = 0; l < L; ++l) {
        const float* hin  = (l == 0) ? x : buf[(l - 1) & 1];
        float*       hout = (l == L - 1) ? (float*)d_out : buf[l & 1];

        zero_kernel<<<zeroBlocks, 256, 0, stream>>>((float4*)aggr, nAggr / 4);
        scatter_kernel<<<scatBlocks, 256, 0, stream>>>(hin, ei, ew, aggr, E);
        gemm_tanh_kernel<<<gemmBlocks, 256, 0, stream>>>(
            aggr, hin, Wfrag + (size_t)l * LAYER_FRAG_ELEMS, b + (size_t)l * CCH, hout, N);
    }
}
